// EncoderRNN_10943576670793
// MI455X (gfx1250) — matrix operand order repaired and checked by reading, not compiled, not run on hardware
//
#include <hip/hip_runtime.h>
#include <hip/hip_bf16.h>

typedef __attribute__((ext_vector_type(16))) __bf16 v16bf;
typedef __attribute__((ext_vector_type(8)))  float  v8f;

#define SEQ_LEN 24
#define HSZ     2048
#define G4      (4 * HSZ)   // 8192

__device__ __forceinline__ unsigned short f2bf(float f) {
    unsigned int u = __float_as_uint(f);
    unsigned int r = u + 0x7FFFu + ((u >> 16) & 1u);  // round-to-nearest-even
    return (unsigned short)(r >> 16);
}

// ---------------- fp32 -> bf16 conversion (HBM-bound, once per call) ----------------
__global__ void cvt_bf16_kernel(const float* __restrict__ src,
                                unsigned short* __restrict__ dst, int n4) {
    int i = blockIdx.x * blockDim.x + threadIdx.x;
    int stride = gridDim.x * blockDim.x;
    for (; i < n4; i += stride) {
        float4 f = ((const float4*)src)[i];
        ushort4 o;
        o.x = f2bf(f.x); o.y = f2bf(f.y); o.z = f2bf(f.z); o.w = f2bf(f.w);
        ((ushort4*)dst)[i] = o;
    }
}

// ---------------- init h0 (bf16) and c0 (f32) ----------------
__global__ void lstm_init(const float* __restrict__ hn, const float* __restrict__ cn,
                          unsigned short* __restrict__ h0, float* __restrict__ c0) {
    int i = blockIdx.x * blockDim.x + threadIdx.x;
    if (i < HSZ) { h0[i] = f2bf(hn[i]); c0[i] = cn[i]; }
}

// ---------------- xg = x @ w_ih^T + b_ih + b_hh  (WMMA GEMM) ----------------
// One wave per (mtile, ntile): mtile in {0,1} covers T=24 padded to 32,
// ntile in [0,512) covers the 8192 gate rows. K = 2048 looped in chunks of 32.
__global__ void lstm_xg_gemm(const unsigned short* __restrict__ xbf,  // [24,2048] bf16
                             const unsigned short* __restrict__ wih,  // [8192,2048] bf16
                             const float* __restrict__ bih,
                             const float* __restrict__ bhh,
                             float* __restrict__ xg) {                // [24,8192] f32
    int wave    = (blockIdx.x * blockDim.x + threadIdx.x) >> 5;   // 0..1023
    int lane    = threadIdx.x & 31;
    int ntile   = wave & 511;
    int mtile   = wave >> 9;
    int rowlane = lane & 15;
    int lh      = lane >> 4;       // half-wave: K-phase selector
    int gbase   = ntile * 16;
    int t       = mtile * 16 + rowlane;

    // B: lane = gate row (N), 16 contiguous bf16 K values (lanes 16-31 get K+16)
    const unsigned short* wrow = wih + (size_t)(gbase + rowlane) * 2048 + lh * 16;
    const unsigned short* xrow = xbf + (size_t)(t < SEQ_LEN ? t : 0) * 2048;
    bool tvalid = (t < SEQ_LEN);

    v8f acc = {};
    // Software pipeline: keep next B tile in flight while WMMA consumes current.
    v16bf Bcur = *(const v16bf*)(wrow);
    for (int kb = 0; kb < 2048; kb += 32) {
        // A tile: M = t (row of x), K contiguous per half-wave (k = lh*16 + s), same order as B
        union { v16bf v; uint4 q[2]; } A;
        uint4 q0 = make_uint4(0, 0, 0, 0), q1 = q0;
        if (tvalid) {
            q0 = *(const uint4*)(xrow + kb + lh * 16);       // K = kb+lh*16 .. +8
            q1 = *(const uint4*)(xrow + kb + lh * 16 + 8);   // K = kb+lh*16+8 .. +8
        }
        A.q[0] = q0; A.q[1] = q1;
        // Next-iteration B (last iter reads 32B past row end -> still inside ws, discarded)
        v16bf Bnext = *(const v16bf*)(wrow + kb + 32);
        __builtin_prefetch(wrow + kb + 96, 0, 1);
        acc = __builtin_amdgcn_wmma_f32_16x16x32_bf16(false, A.v, false, Bcur,
                                                      (short)0, acc, false, false);
        Bcur = Bnext;
    }

    // D layout: M = r + (lane<16 ? 0 : 8), N = lane%16
    int g = gbase + rowlane;
    float bias = bih[g] + bhh[g];
    #pragma unroll
    for (int r = 0; r < 8; ++r) {
        int tt = mtile * 16 + r + lh * 8;
        if (tt < SEQ_LEN) xg[(size_t)tt * G4 + g] = acc[r] + bias;
    }
}

// ---------------- one recurrence step: gates = xg[t] + h @ w_hh^T; cell update ----
// 128 blocks x 256 threads. Block = one 16-wide j tile.
// Wave w in [0,8): gate block (w&3) of {i,f,g,o}, K half (w>>2). 32 WMMA iters each.
__global__ void lstm_step(const unsigned short* __restrict__ whh,   // [8192,2048] bf16
                          const float* __restrict__ xg,             // [24,8192]
                          const unsigned short* __restrict__ h_in,  // [2048] bf16
                          const float* __restrict__ c_in,           // [2048] f32
                          unsigned short* __restrict__ h_out,
                          float* __restrict__ c_out,
                          float* __restrict__ out_h,                // d_out + t*2048
                          int t) {
    __shared__ float part[8][16];
    int wg      = blockIdx.x;            // j tile: [wg*16, wg*16+16)
    int wid     = threadIdx.x >> 5;      // 0..7
    int lane    = threadIdx.x & 31;
    int rowlane = lane & 15;
    int lh      = lane >> 4;
    int gb      = wid & 3;               // gate block
    int khalf   = wid >> 2;              // K half
    int gbase   = gb * HSZ + wg * 16;

    const unsigned short* wrow = whh + (size_t)(gbase + rowlane) * 2048 + lh * 16;

    v8f acc = {};
    int k0 = khalf * 1024;
    v16bf Bcur = *(const v16bf*)(wrow + k0);
    for (int kb = k0; kb < k0 + 1024; kb += 32) {
        // A: only M=0 row carries h (mat-vec); lanes 0 and 16 load, rest zero.
        union { v16bf v; uint4 q[2]; } A;
        uint4 q0 = make_uint4(0, 0, 0, 0), q1 = q0;
        if (rowlane == 0) {
            q0 = *(const uint4*)(h_in + kb + lh * 16);       // K = kb+lh*16 .. +8
            q1 = *(const uint4*)(h_in + kb + lh * 16 + 8);   // K = kb+lh*16+8 .. +8
        }
        A.q[0] = q0; A.q[1] = q1;
        // Double-buffered B keeps one load in flight across the WMMA (hides L2 latency
        // on the serial h-chain). Last iteration reads 32B past the row -> inside ws.
        v16bf Bnext = *(const v16bf*)(wrow + kb + 32);
        __builtin_prefetch(wrow + kb + 96, 0, 1);
        acc = __builtin_amdgcn_wmma_f32_16x16x32_bf16(false, A.v, false, Bcur,
                                                      (short)0, acc, false, false);
        Bcur = Bnext;
    }

    // Row M=0 of D lives in acc[0] of lanes 0..15 (N = lane)
    if (lane < 16) part[wid][lane] = acc[0];
    __syncthreads();

    if (threadIdx.x < 16) {
        int n = threadIdx.x;
        int j = wg * 16 + n;
        const float* xgt = xg + (size_t)t * G4;
        float gi = part[0][n] + part[4][n] + xgt[0 * HSZ + j];
        float gf = part[1][n] + part[5][n] + xgt[1 * HSZ + j];
        float gg = part[2][n] + part[6][n] + xgt[2 * HSZ + j];
        float go = part[3][n] + part[7][n] + xgt[3 * HSZ + j];
        float i_ = 1.0f / (1.0f + __expf(-gi));
        float f_ = 1.0f / (1.0f + __expf(-gf));
        float g_ = tanhf(gg);
        float o_ = 1.0f / (1.0f + __expf(-go));
        float c  = f_ * c_in[j] + i_ * g_;
        float h  = o_ * tanhf(c);
        c_out[j] = c;
        out_h[j] = h;
        h_out[j] = f2bf(h);
    }
}

// ---------------- tail: hT = out[23], cT = final c ----------------
__global__ void lstm_tail(float* __restrict__ out, const float* __restrict__ c_fin) {
    int i = blockIdx.x * blockDim.x + threadIdx.x;
    if (i < HSZ) {
        out[SEQ_LEN * HSZ + i]       = out[(SEQ_LEN - 1) * HSZ + i]; // hT
        out[(SEQ_LEN + 1) * HSZ + i] = c_fin[i];                     // cT
    }
}

extern "C" void kernel_launch(void* const* d_in, const int* in_sizes, int n_in,
                              void* d_out, int out_size, void* d_ws, size_t ws_size,
                              hipStream_t stream) {
    const float* x    = (const float*)d_in[0];  // [24,1,2048]
    const float* hn   = (const float*)d_in[1];  // [1,1,2048]
    const float* cn   = (const float*)d_in[2];  // [1,1,2048]
    const float* w_ih = (const float*)d_in[3];  // [8192,2048]
    const float* w_hh = (const float*)d_in[4];  // [8192,2048]
    const float* b_ih = (const float*)d_in[5];  // [8192]
    const float* b_hh = (const float*)d_in[6];  // [8192]
    float* out = (float*)d_out;                 // 24*2048 (out) + 2048 (hT) + 2048 (cT)

    // Workspace layout (bytes)
    const size_t W_ELEMS = (size_t)G4 * 2048;            // 16,777,216 per weight matrix
    char* ws = (char*)d_ws;
    unsigned short* wih_bf = (unsigned short*)(ws);                        // 32 MB
    unsigned short* whh_bf = (unsigned short*)(ws + W_ELEMS * 2);          // 32 MB
    float*          xg     = (float*)(ws + W_ELEMS * 4);                   // 768 KB
    unsigned short* xbf    = (unsigned short*)(ws + W_ELEMS * 4 + (size_t)SEQ_LEN * G4 * 4); // 96 KB
    unsigned short* hbuf   = (unsigned short*)((char*)xbf + (size_t)SEQ_LEN * HSZ * 2);      // [2][2048] bf16
    float*          cbuf   = (float*)((char*)hbuf + 2 * HSZ * 2);          // [2][2048] f32

    // 1. Convert weights + x to bf16 (reads ~128 MB, writes ~64 MB: ~8 us at 23.3 TB/s)
    cvt_bf16_kernel<<<1024, 256, 0, stream>>>(w_ih, wih_bf, (int)(W_ELEMS / 4));
    cvt_bf16_kernel<<<1024, 256, 0, stream>>>(w_hh, whh_bf, (int)(W_ELEMS / 4));
    cvt_bf16_kernel<<<48, 256, 0, stream>>>(x, xbf, SEQ_LEN * HSZ / 4);

    // 2. Initialize h0 (bf16) / c0 into buffer 0
    lstm_init<<<8, 256, 0, stream>>>(hn, cn, hbuf, cbuf);

    // 3. Input-side gate pre-activations for all timesteps (WMMA GEMM)
    lstm_xg_gemm<<<256, 128, 0, stream>>>(xbf, wih_bf, b_ih, b_hh, xg);

    // 4. Serial recurrence: one launch per step (kernel boundary = grid-wide sync),
    //    h/c double-buffered so intra-step readers never race writers.
    for (int t = 0; t < SEQ_LEN; ++t) {
        int rb = t & 1, wb = rb ^ 1;
        lstm_step<<<128, 256, 0, stream>>>(whh_bf, xg,
                                           hbuf + rb * HSZ, cbuf + rb * HSZ,
                                           hbuf + wb * HSZ, cbuf + wb * HSZ,
                                           out + (size_t)t * HSZ, t);
    }

    // 5. hT / cT tuple tail (final c is in buffer 0 after 24 steps)
    lstm_tail<<<8, 256, 0, stream>>>(out, cbuf);
}
